// TextCnnWithFusion_6811818131553
// MI455X (gfx1250) — matrix operand order repaired and checked by reading, not compiled, not run on hardware
//
#include <hip/hip_runtime.h>

typedef __attribute__((ext_vector_type(16))) __bf16 v16bf;
typedef __attribute__((ext_vector_type(8)))  float  v8f;

#define B_   64
#define L_   800
#define DP   128      // D=100 padded to 128 (K multiple of 32)
#define NFP  64       // 50 filters padded to 64

union Frag16 { uint4 u[2]; v16bf v; };

static __device__ __forceinline__ v8f wmma_bf16(v16bf a, v16bf b, v8f c) {
  return __builtin_amdgcn_wmma_f32_16x16x32_bf16(false, a, false, b, (short)0, c, false, false);
}

struct Ptrs8 { const float* p[8]; };

// Async-copy one contiguous 8KB block (32 rows x 256B) global -> LDS.
// 16 instructions x 32 lanes x 16B. Tracked by ASYNCcnt (in-order retire).
static __device__ __forceinline__ void async_copy_8k(const __bf16* gsrc,
                                                     unsigned lds_base, int lane) {
  unsigned lds = lds_base + lane * 16;
  unsigned long long ga = (unsigned long long)((const char*)gsrc + lane * 16);
#pragma unroll
  for (int i = 0; i < 16; ++i) {
    asm volatile("global_load_async_to_lds_b128 %0, %1, off"
                 :: "v"(lds), "v"(ga) : "memory");
    lds += 512;
    ga  += 512;
  }
}

// ---------------------------------------------------------------------------
// 1) Embedding gather -> xe_bf16 [B][L][128] and transposed xeT_bf16 [B][128][L]
// ---------------------------------------------------------------------------
__global__ void gather_kernel(const int* __restrict__ x, const float* __restrict__ emb,
                              __bf16* __restrict__ xe, __bf16* __restrict__ xeT) {
  int idx = blockIdx.x * 256 + threadIdx.x;       // total = 64*800*128 = 6,553,600
  int d  = idx & 127;
  int bl = idx >> 7;                              // b*800 + l
  int l  = bl % L_;
  int b  = bl / L_;
  int tok = x[bl];
  float v = (d < 100) ? emb[tok * 100 + d] : 0.f;
  __bf16 bv = (__bf16)v;
  xe[idx] = bv;
  xeT[((size_t)b * DP + d) * L_ + l] = bv;
}

// ---------------------------------------------------------------------------
// 2) Weight prep: cw{br}_{si} [50,1,s,100] -> wpad [br][64][tap(11)][128] bf16
// ---------------------------------------------------------------------------
__global__ void wprep_kernel(Ptrs8 wp, __bf16* __restrict__ wpad) {
  int idx = blockIdx.x * 256 + threadIdx.x;       // total = 2*64*11*128 = 180,224
  int d    = idx & 127;
  int rest = idx >> 7;
  int tap  = rest % 11;
  int f    = (rest / 11) & 63;
  int br   = rest / (11 * 64);
  const int CUM[5] = {0, 1, 3, 6, 11};
  const int SZ[4]  = {1, 2, 3, 5};
  int si = 0;
  while (tap >= CUM[si + 1]) si++;
  int t = tap - CUM[si];
  float v = 0.f;
  if (f < 50 && d < 100) v = wp.p[br * 4 + si][(f * SZ[si] + t) * 100 + d];
  wpad[idx] = (__bf16)v;
}

// ---------------------------------------------------------------------------
// 3) Fused attention: per (b, 16-row tile), flash-style, never materializes s.
//    Key-row tiles are double-buffered into LDS with ASYNCcnt-tracked copies.
//    x_hat = (exp(xe.xe^T)*valid / rowsum) @ xe + xe,  valid = offdiag & mask
// ---------------------------------------------------------------------------
__global__ void __launch_bounds__(32)
attn_kernel(const __bf16* __restrict__ xe, const __bf16* __restrict__ xeT,
            const int* __restrict__ mask, __bf16* __restrict__ xhat) {
  __shared__ alignas(16) __bf16 P[16 * 32];         // exp-weights tile, 1KB
  __shared__ alignas(16) __bf16 Xrow[2][32 * DP];   // double-buffered key tiles, 2x8KB

  const int lane = threadIdx.x;
  const int half = lane >> 4, lrow = lane & 15;
  const int i0 = blockIdx.x * 16;
  const int b  = blockIdx.y;

  const __bf16* xeB  = xe  + (size_t)b * L_ * DP;
  const __bf16* xeTB = xeT + (size_t)b * DP * L_;
  const int*    mB   = mask + b * L_;

  unsigned xbase[2] = { (unsigned)(uintptr_t)&Xrow[0][0],
                        (unsigned)(uintptr_t)&Xrow[1][0] };

  // A fragments for the 16 query rows (reused for every column tile)
  v16bf afr[4];
#pragma unroll
  for (int kc = 0; kc < 4; ++kc) {
    const __bf16* p = xeB + (i0 + lrow) * DP + kc * 32;
    Frag16 f;
    f.u[0] = *(const uint4*)(p + half * 16);
    f.u[1] = *(const uint4*)(p + half * 16 + 8);
    afr[kc] = f.v;
  }

  int mrow[8];
#pragma unroll
  for (int r = 0; r < 8; ++r) mrow[r] = mB[i0 + half * 8 + r];

  float rowsum[8];
  v8f o[8];
#pragma unroll
  for (int r = 0; r < 8; ++r) rowsum[r] = 0.f;
#pragma unroll
  for (int t = 0; t < 8; ++t) o[t] = (v8f){0,0,0,0,0,0,0,0};

  async_copy_8k(xeB, xbase[0], lane);               // prefetch tile jt=0

  for (int jt = 0; jt < L_ / 32; ++jt) {
    const int j0 = jt * 32;
    const int cb = jt & 1;
    if (jt + 1 < L_ / 32) {                         // kick off next tile, then wait
      async_copy_8k(xeB + (size_t)(j0 + 32) * DP, xbase[cb ^ 1], lane);
      asm volatile("s_wait_asynccnt 0x10" ::: "memory");  // older 16 done (in-order)
    } else {
      asm volatile("s_wait_asynccnt 0x0" ::: "memory");
    }
    __syncthreads();

    const __bf16* xr = &Xrow[cb][0];
    v8f c0 = (v8f){0,0,0,0,0,0,0,0};
    v8f c1 = (v8f){0,0,0,0,0,0,0,0};
#pragma unroll
    for (int kc = 0; kc < 4; ++kc) {                // scores: K = 128 dims, B from LDS
      const __bf16* p0 = xr + lrow * DP + kc * 32 + half * 16;
      Frag16 fb0; fb0.u[0] = *(const uint4*)p0; fb0.u[1] = *(const uint4*)(p0 + 8);
      c0 = wmma_bf16(afr[kc], fb0.v, c0);
      const __bf16* p1 = xr + (16 + lrow) * DP + kc * 32 + half * 16;
      Frag16 fb1; fb1.u[0] = *(const uint4*)p1; fb1.u[1] = *(const uint4*)(p1 + 8);
      c1 = wmma_bf16(afr[kc], fb1.v, c1);
    }
    const int mc0 = mB[j0 + lrow];
    const int mc1 = mB[j0 + 16 + lrow];
#pragma unroll
    for (int r = 0; r < 8; ++r) {                   // exp + validity, C layout (r, lane)
      const int row = i0 + half * 8 + r;
      float w0 = (mrow[r] && mc0 && row != (j0 + lrow))      ? __expf(c0[r]) : 0.f;
      float w1 = (mrow[r] && mc1 && row != (j0 + 16 + lrow)) ? __expf(c1[r]) : 0.f;
      rowsum[r] += w0 + w1;
      P[(half * 8 + r) * 32 + lrow]      = (__bf16)w0;
      P[(half * 8 + r) * 32 + lrow + 16] = (__bf16)w1;
    }
    __syncthreads();
    Frag16 pf;                                      // P tile -> A-fragment layout
    const __bf16* pp = P + lrow * 32;
    pf.u[0] = *(const uint4*)(pp + half * 16);
    pf.u[1] = *(const uint4*)(pp + half * 16 + 8);
    v16bf pv = pf.v;
#pragma unroll
    for (int t = 0; t < 8; ++t) {                   // apply: O(16x128) += P(16x32) @ X(32x128)
      const __bf16* bp = xeTB + (t * 16 + lrow) * L_ + j0 + half * 16;
      Frag16 fb; fb.u[0] = *(const uint4*)bp; fb.u[1] = *(const uint4*)(bp + 8);
      o[t] = wmma_bf16(pv, fb.v, o[t]);
    }
    __syncthreads();
  }

  float inv[8];
#pragma unroll
  for (int r = 0; r < 8; ++r) {                     // reduce over 16 lanes of each half
    float s = rowsum[r];
    s += __shfl_xor(s, 1, 32);
    s += __shfl_xor(s, 2, 32);
    s += __shfl_xor(s, 4, 32);
    s += __shfl_xor(s, 8, 32);
    inv[r] = 1.f / (s + 1e-13f);
  }
#pragma unroll
  for (int t = 0; t < 8; ++t) {
#pragma unroll
    for (int r = 0; r < 8; ++r) {
      const int row = i0 + half * 8 + r;
      const int dim = t * 16 + lrow;
      float val = o[t][r] * inv[r] + (float)xeB[row * DP + dim];   // residual
      xhat[((size_t)b * L_ + row) * DP + dim] = (__bf16)val;
    }
  }
}

// ---------------------------------------------------------------------------
// 4) Conv-as-GEMM + fused ReLU + global-max-pool (bitwise atomic max, vals>=0)
// ---------------------------------------------------------------------------
__global__ void __launch_bounds__(32)
conv_kernel(const __bf16* __restrict__ xe, const __bf16* __restrict__ xhat,
            const __bf16* __restrict__ wpad, Ptrs8 bp, float* __restrict__ pooled) {
  const int lane = threadIdx.x;
  const int half = lane >> 4, lrow = lane & 15;
  const int ftile = blockIdx.x & 3;
  const int ptile = blockIdx.x >> 2;
  const int si = blockIdx.y;
  const int b  = blockIdx.z >> 1;
  const int br = blockIdx.z & 1;
  const int SZ[4]  = {1, 2, 3, 5};
  const int CUM[4] = {0, 1, 3, 6};
  const int s = SZ[si];
  const int p0 = ptile * 16, fbase = ftile * 16;
  const int Ps = L_ - s + 1;

  const __bf16* src  = (br == 0 ? xe : xhat) + (size_t)b * L_ * DP;
  const __bf16* wrow = wpad + ((size_t)(br * NFP + fbase + lrow) * 11 + CUM[si]) * DP;

  v8f c = (v8f){0,0,0,0,0,0,0,0};
  for (int kk = 0; kk < s * DP; kk += 32) {         // K = taps * 128 padded dims
    const int tap = kk >> 7, dd = kk & 127;
    const int row = p0 + lrow + tap;
    Frag16 fa;
    if (row < L_) {
      const __bf16* p = src + row * DP + dd;
      fa.u[0] = *(const uint4*)(p + half * 16);
      fa.u[1] = *(const uint4*)(p + half * 16 + 8);
    } else {
      fa.u[0] = make_uint4(0, 0, 0, 0);
      fa.u[1] = make_uint4(0, 0, 0, 0);
    }
    Frag16 fw;
    const __bf16* wpk = wrow + kk + half * 16;
    fw.u[0] = *(const uint4*)wpk;
    fw.u[1] = *(const uint4*)(wpk + 8);
    c = wmma_bf16(fa.v, fw.v, c);
  }

  const int f = fbase + lrow;                       // this lane's filter column
  const float bias = (f < 50) ? bp.p[br * 4 + si][f] : 0.f;
  float m8 = 0.f;
#pragma unroll
  for (int r = 0; r < 8; ++r) {
    const int pos = p0 + half * 8 + r;
    float v = fmaxf(c[r] + bias, 0.f);
    if (pos >= Ps) v = 0.f;                         // OOB positions contribute relu-floor 0
    m8 = fmaxf(m8, v);
  }
  float m = fmaxf(m8, __shfl_xor(m8, 16, 32));      // combine the two row-halves per column
  if (lane < 16) {
    const int ff = fbase + lane;
    if (ff < 50)
      atomicMax((unsigned int*)&pooled[((size_t)br * B_ + b) * 200 + si * 50 + ff],
                __float_as_uint(m));                // bit-order == float-order for vals >= 0
  }
}

// ---------------------------------------------------------------------------
// 5) Final FC: out[b][o] = pooled1[b]·fw1[o] + fb1[o] + pooled2[b]·fw2[o] + fb2[o]
// ---------------------------------------------------------------------------
__global__ void fc_kernel(const float* __restrict__ pooled,
                          const float* __restrict__ fw1, const float* __restrict__ fb1,
                          const float* __restrict__ fw2, const float* __restrict__ fb2,
                          float* __restrict__ out) {
  int idx = blockIdx.x * 256 + threadIdx.x;
  if (idx >= B_ * 50) return;
  int b = idx / 50, o = idx % 50;
  float acc = fb1[o] + fb2[o];
  const float* p1 = pooled + (size_t)b * 200;
  const float* p2 = pooled + ((size_t)B_ + b) * 200;
  for (int k = 0; k < 200; ++k)
    acc += p1[k] * fw1[o * 200 + k] + p2[k] * fw2[o * 200 + k];
  out[idx] = acc;
}

// ---------------------------------------------------------------------------
extern "C" void kernel_launch(void* const* d_in, const int* in_sizes, int n_in,
                              void* d_out, int out_size, void* d_ws, size_t ws_size,
                              hipStream_t stream) {
  const int*   x    = (const int*)d_in[0];
  const int*   mask = (const int*)d_in[2];
  const float* emb  = (const float*)d_in[3];
  const float* fw1  = (const float*)d_in[12];
  const float* fb1  = (const float*)d_in[13];
  const float* fw2  = (const float*)d_in[22];
  const float* fb2  = (const float*)d_in[23];

  Ptrs8 wps = {{ (const float*)d_in[4],  (const float*)d_in[6],
                 (const float*)d_in[8],  (const float*)d_in[10],
                 (const float*)d_in[14], (const float*)d_in[16],
                 (const float*)d_in[18], (const float*)d_in[20] }};
  Ptrs8 bps = {{ (const float*)d_in[5],  (const float*)d_in[7],
                 (const float*)d_in[9],  (const float*)d_in[11],
                 (const float*)d_in[15], (const float*)d_in[17],
                 (const float*)d_in[19], (const float*)d_in[21] }};

  const size_t XE_BYTES   = (size_t)B_ * L_ * DP * 2;   // 13,107,200
  const size_t WPAD_BYTES = (size_t)2 * NFP * 11 * DP * 2;
  const size_t POOL_BYTES = (size_t)2 * B_ * 200 * sizeof(float);

  char*   ws     = (char*)d_ws;
  __bf16* xe     = (__bf16*)(ws);
  __bf16* xeT    = (__bf16*)(ws + XE_BYTES);
  __bf16* xhat   = (__bf16*)(ws + 2 * XE_BYTES);
  __bf16* wpad   = (__bf16*)(ws + 3 * XE_BYTES);
  float*  pooled = (float*) (ws + 3 * XE_BYTES + WPAD_BYTES);

  gather_kernel<<<(B_ * L_ * DP) / 256, 256, 0, stream>>>(x, emb, xe, xeT);
  wprep_kernel<<<(2 * NFP * 11 * DP) / 256, 256, 0, stream>>>(wps, wpad);
  hipMemsetAsync(pooled, 0, POOL_BYTES, stream);

  attn_kernel<<<dim3(L_ / 16, B_), 32, 0, stream>>>(xe, xeT, mask, xhat);
  conv_kernel<<<dim3(50 * 4, 4, B_ * 2), 32, 0, stream>>>(xe, xhat, wpad, bps, pooled);
  fc_kernel<<<(B_ * 50 + 255) / 256, 256, 0, stream>>>(pooled, fw1, fb1, fw2, fb2,
                                                       (float*)d_out);
}
